// Attention_74182675136800
// MI455X (gfx1250) — matrix operand order repaired and checked by reading, not compiled, not run on hardware
//
#include <hip/hip_runtime.h>
#include <hip/hip_bf16.h>

typedef _Float16 half_t;
typedef __attribute__((ext_vector_type(8)))  _Float16 v8h;
typedef __attribute__((ext_vector_type(16))) _Float16 v16h;
typedef __attribute__((ext_vector_type(8)))  float    v8f;

#define B_ 2
#define T_ 2048
#define D_ 1024
#define H_ 16
#define HD_ 64

// ---------------------------------------------------------------------------
// WMMA wrapper: D = A(16x32 f16) * B(32x16 f16) + C(16x16 f32)
// ---------------------------------------------------------------------------
__device__ __forceinline__ v8f wmma_f16(v16h a, v16h b, v8f c) {
  return __builtin_amdgcn_wmma_f32_16x16x32_f16(
      /*neg_a=*/false, a, /*neg_b=*/false, b,
      /*c_mod=*/(short)0, c, /*reuse_a=*/false, /*reuse_b=*/false);
}

// A fragment (16x32, MxK): src row-major [*, ld] f16. Row = lane%16.
// Contiguous k order used here (same as B frags): half h -> K = (lane/16)*16 + h
// => two contiguous 8-half (16B) chunks per lane.
__device__ __forceinline__ v16h load_a_frag(const half_t* base, int ld, int m0, int k0) {
  int lane = threadIdx.x & 31;
  int row  = m0 + (lane & 15);
  int hi   = lane >> 4;
  const half_t* p = base + (size_t)row * ld + k0 + hi * 16;
  v8h lo = *(const v8h*)p;
  v8h hh = *(const v8h*)(p + 8);
  v16h r;
#pragma unroll
  for (int i = 0; i < 8; ++i) { r[i] = lo[i]; r[8 + i] = hh[i]; }
  return r;
}

// B fragment (32x16, KxN): src stored K-major per column: base[col*ld + k].
// Contiguous k order used here (same as A frags): half h -> K = (lane/16)*16 + h  => one contiguous 16-half (32B) chunk.
__device__ __forceinline__ v16h load_b_frag(const half_t* base, int ld, int n0, int k0) {
  int lane = threadIdx.x & 31;
  int col  = n0 + (lane & 15);
  int hi   = lane >> 4;
  return *(const v16h*)(base + (size_t)col * ld + k0 + hi * 16);
}

// ---------------------------------------------------------------------------
// Prep kernels
// ---------------------------------------------------------------------------
__global__ void k_cvt_f16(const float* __restrict__ src, half_t* __restrict__ dst, int n) {
  int i = blockIdx.x * blockDim.x + threadIdx.x;
  if (i < n) dst[i] = (half_t)src[i];
}

// dst[n][k] = (f16) src[k][n]   (weight transpose -> K-major columns for B frags)
__global__ void k_transpose_f16(const float* __restrict__ src, half_t* __restrict__ dst,
                                int rows /*k*/, int cols /*n*/) {
  int i = blockIdx.x * blockDim.x + threadIdx.x;
  if (i < rows * cols) {
    int k = i / cols, n = i % cols;
    dst[(size_t)n * rows + k] = (half_t)src[i];
  }
}

// ---------------------------------------------------------------------------
// GEMM: C[M,N] = A[M,K] * W[K,N], A row-major f16, W given transposed (WT[n][k]).
// One wave (block of 32) per 64x64 output tile: 4 M-frags x 4 N-frags,
// 16 f32 accumulators (128 VGPRs). Per 32-K step: 16 x B128 loads feed
// 16 WMMAs (1.0 loads/WMMA). __launch_bounds__(32,1) -> full VGPR budget,
// no accumulator spills.
// mode 0: store f16 per-head layout [b][h][t][64]     (Q, K)
// mode 1: store f16 transposed per-head [b][h][64][t] (V)
// mode 2: store f32 row-major [M,N]                   (final output)
// ---------------------------------------------------------------------------
__global__ void __launch_bounds__(32, 1)
k_gemm(const half_t* __restrict__ A, const half_t* __restrict__ WT,
       half_t* __restrict__ out_h, float* __restrict__ out_f,
       int M, int Kdim, int N, int mode) {
  int ntn = N / 64;
  int m0  = (blockIdx.x / ntn) * 64;
  int n0  = (blockIdx.x % ntn) * 64;

  v8f acc[4][4] = {};   // [mt][nt]
  for (int k0 = 0; k0 < Kdim; k0 += 32) {
    v16h a[4];
#pragma unroll
    for (int mt = 0; mt < 4; ++mt) a[mt] = load_a_frag(A, Kdim, m0 + mt * 16, k0);
#pragma unroll
    for (int nt = 0; nt < 4; ++nt) {
      v16h b = load_b_frag(WT, Kdim, n0 + nt * 16, k0);
#pragma unroll
      for (int mt = 0; mt < 4; ++mt)
        acc[mt][nt] = wmma_f16(a[mt], b, acc[mt][nt]);
    }
  }

  int lane = threadIdx.x & 31, hi = lane >> 4, nl = lane & 15;
#pragma unroll
  for (int mt = 0; mt < 4; ++mt) {
#pragma unroll
    for (int nt = 0; nt < 4; ++nt) {
#pragma unroll
      for (int r = 0; r < 8; ++r) {
        int m = m0 + mt * 16 + r + 8 * hi;
        int n = n0 + nt * 16 + nl;
        float v = acc[mt][nt][r];
        if (mode == 2) {
          out_f[(size_t)m * N + n] = v;
        } else {
          int b = m / T_, t = m % T_, h = n >> 6, d = n & 63;
          if (mode == 0)
            out_h[((((size_t)b * H_ + h) * T_) + t) * HD_ + d] = (half_t)v;
          else
            out_h[((((size_t)b * H_ + h) * HD_) + d) * T_ + t] = (half_t)v;
        }
      }
    }
  }
}

// ---------------------------------------------------------------------------
// Flash attention, causal. One wave per (b,h, 16-query tile).
// Q,K: [b][h][t][64] f16.  VT: [b][h][64][t] f16.  attnh out: [b][t][D] f16.
// ---------------------------------------------------------------------------
__global__ void __launch_bounds__(32, 1)
k_flash_attn(const half_t* __restrict__ Qh, const half_t* __restrict__ Kh,
             const half_t* __restrict__ VT, half_t* __restrict__ attnh) {
  __shared__ __align__(32) half_t lds[16 * 32];

  int qtile = blockIdx.x;        // 0 .. T/16-1
  int bh    = blockIdx.y;        // 0 .. B*H-1
  int q0    = qtile * 16;

  const half_t* Qb = Qh + (size_t)bh * T_ * HD_;
  const half_t* Kb = Kh + (size_t)bh * T_ * HD_;
  const half_t* Vb = VT + (size_t)bh * HD_ * T_;

  int lane = threadIdx.x & 31, hi = lane >> 4, nl = lane & 15;

  v16h qa0 = load_a_frag(Qb, HD_, q0, 0);    // d = 0..31
  v16h qa1 = load_a_frag(Qb, HD_, q0, 32);   // d = 32..63

  v8f  O[4] = {};
  float rowM[8], rowL[8];
#pragma unroll
  for (int r = 0; r < 8; ++r) { rowM[r] = -3.0e38f; rowL[r] = 0.0f; }

  const float sscale = 0.125f;   // 1/sqrt(64)
  int nk = q0 / 32 + 1;          // covers keys 0 .. q0+15 (mask trims the rest)

  for (int ks = 0; ks < nk; ++ks) {
    int k0 = ks * 32;

    // ---- scores S[16 x 32] = Q . K^T  (contraction over d; K rows are K-major)
    v16h kb0a = load_b_frag(Kb, HD_, k0,      0);
    v16h kb0b = load_b_frag(Kb, HD_, k0,      32);
    v16h kb1a = load_b_frag(Kb, HD_, k0 + 16, 0);
    v16h kb1b = load_b_frag(Kb, HD_, k0 + 16, 32);
    v8f Sa = {}; Sa = wmma_f16(qa0, kb0a, Sa); Sa = wmma_f16(qa1, kb0b, Sa);
    v8f Sb = {}; Sb = wmma_f16(qa0, kb1a, Sb); Sb = wmma_f16(qa1, kb1b, Sb);

    // ---- causal mask + online softmax (row = r + 8*hi, col = nl / 16+nl)
#pragma unroll
    for (int r = 0; r < 8; ++r) {
      int q  = q0 + r + 8 * hi;
      float sa = Sa[r] * sscale;
      float sb = Sb[r] * sscale;
      if (k0 + nl > q)      sa = -1.0e30f;
      if (k0 + 16 + nl > q) sb = -1.0e30f;

      float mx = fmaxf(sa, sb);
#pragma unroll
      for (int m = 1; m < 16; m <<= 1) mx = fmaxf(mx, __shfl_xor(mx, m, 32));
      float mnew  = fmaxf(rowM[r], mx);
      float scale = __expf(rowM[r] - mnew);
      rowM[r] = mnew;

      float pa = __expf(sa - mnew);
      float pb = __expf(sb - mnew);
      float rs = pa + pb;
#pragma unroll
      for (int m = 1; m < 16; m <<= 1) rs += __shfl_xor(rs, m, 32);
      rowL[r] = rowL[r] * scale + rs;
#pragma unroll
      for (int dt = 0; dt < 4; ++dt) O[dt][r] *= scale;

      int row = r + 8 * hi;
      lds[row * 32 + nl]      = (half_t)pa;
      lds[row * 32 + 16 + nl] = (half_t)pb;
    }
    __syncthreads();

    // ---- regather P (16x32) into A-fragment layout
    v16h pf;
#pragma unroll
    for (int h = 0; h < 16; ++h) {
      int kk = hi * 16 + h;
      pf[h] = lds[nl * 32 + kk];
    }
    __syncthreads();

    // ---- O += P . V   (contraction over kk; VT columns are kk-major)
#pragma unroll
    for (int dt = 0; dt < 4; ++dt) {
      int d = dt * 16 + nl;
      v16h vf = *(const v16h*)(Vb + (size_t)d * T_ + k0 + hi * 16);
      O[dt] = wmma_f16(pf, vf, O[dt]);
    }
  }

  // ---- epilogue: normalize, store f16 [b][t][h*64+d]
  int b = bh / H_, h = bh % H_;
#pragma unroll
  for (int dt = 0; dt < 4; ++dt) {
#pragma unroll
    for (int r = 0; r < 8; ++r) {
      int q = q0 + r + 8 * hi;
      float v = O[dt][r] / rowL[r];
      attnh[(((size_t)b * T_ + q) * D_) + h * HD_ + dt * 16 + nl] = (half_t)v;
    }
  }
}

// ---------------------------------------------------------------------------
extern "C" void kernel_launch(void* const* d_in, const int* in_sizes, int n_in,
                              void* d_out, int out_size, void* d_ws, size_t ws_size,
                              hipStream_t stream) {
  const float* x  = (const float*)d_in[0];
  const float* Wq = (const float*)d_in[1];
  const float* Wk = (const float*)d_in[2];
  const float* Wv = (const float*)d_in[3];
  const float* Wo = (const float*)d_in[4];
  float* out = (float*)d_out;

  const size_t MT = (size_t)B_ * T_;          // 4096 rows
  const size_t NX = MT * D_;                  // 4M elems
  const size_t NW = (size_t)D_ * D_;          // 1M elems

  half_t* ws   = (half_t*)d_ws;
  half_t* xh   = ws;                 // [B*T][D]
  half_t* WqT  = xh  + NX;           // [D][D] (n-major)
  half_t* WkT  = WqT + NW;
  half_t* WvT  = WkT + NW;
  half_t* WoT  = WvT + NW;
  half_t* Qh   = WoT + NW;           // [b][h][t][64]
  half_t* Kh   = Qh  + NX;
  half_t* VTs  = Kh  + NX;           // [b][h][64][t]
  half_t* atth = VTs + NX;           // [b][t][D]

  // 1) f32 -> f16 conversions / weight transposes
  k_cvt_f16<<<(NX + 255) / 256, 256, 0, stream>>>(x, xh, (int)NX);
  k_transpose_f16<<<(NW + 255) / 256, 256, 0, stream>>>(Wq, WqT, D_, D_);
  k_transpose_f16<<<(NW + 255) / 256, 256, 0, stream>>>(Wk, WkT, D_, D_);
  k_transpose_f16<<<(NW + 255) / 256, 256, 0, stream>>>(Wv, WvT, D_, D_);
  k_transpose_f16<<<(NW + 255) / 256, 256, 0, stream>>>(Wo, WoT, D_, D_);

  // 2) Q/K/V projections (one wave per 64x64 tile)
  int gemm_blocks = (int)((MT / 64) * (D_ / 64));   // 1024
  k_gemm<<<gemm_blocks, 32, 0, stream>>>(xh, WqT, Qh,  nullptr, (int)MT, D_, D_, 0);
  k_gemm<<<gemm_blocks, 32, 0, stream>>>(xh, WkT, Kh,  nullptr, (int)MT, D_, D_, 0);
  k_gemm<<<gemm_blocks, 32, 0, stream>>>(xh, WvT, VTs, nullptr, (int)MT, D_, D_, 1);

  // 3) causal flash attention: one wave per (16-query tile, b*h)
  k_flash_attn<<<dim3(T_ / 16, B_ * H_), 32, 0, stream>>>(Qh, Kh, VTs, atth);

  // 4) output projection -> fp32
  k_gemm<<<gemm_blocks, 32, 0, stream>>>(atth, WoT, nullptr, out, (int)MT, D_, D_, 2);
}
